// APPNP_89696097009666
// MI455X (gfx1250) — matrix operand order repaired and checked by reading, not compiled, not run on hardware
//
#include <hip/hip_runtime.h>
#include <math.h>

// Problem constants (match reference)
#define N_NODES 50000
#define F_IN    1024
#define H_DIM   256
#define C_DIM   64
#define NNZ_CNT 1000000
#define E_CNT   500000
#define ALPHA_F 0.1f
#define K_ITERS 10

typedef __bf16 bf16_t;
typedef __attribute__((ext_vector_type(16))) __bf16 v16bf;
typedef __attribute__((ext_vector_type(8)))  __bf16 v8bf;
typedef __attribute__((ext_vector_type(8)))  float  v8f;

// ---------------------------------------------------------------- utilities
__global__ void k_zero_f32(float* __restrict__ p, long n) {
    long i = (long)blockIdx.x * blockDim.x + threadIdx.x;
    if (i < n) p[i] = 0.0f;
}

// scatter-add sparse features into dense X[N,F] (fp32)
__global__ void k_scatter_feat(const int* __restrict__ rows,
                               const int* __restrict__ cols,
                               const float* __restrict__ vals,
                               float* __restrict__ X, int nnz) {
    int i = blockIdx.x * blockDim.x + threadIdx.x;
    if (i < nnz)
        atomicAdd(&X[(size_t)rows[i] * F_IN + cols[i]], vals[i]);
}

// Pack a row-major fp32 weight W[K, Nd] into the WMMA B-operand bf16 layout:
// flat[((kt*nNT + nt)*32 + lane)*16 + j] = W[kt*32 + (lane>=16?8:0) + (j<8 ? j : j+8)][nt*16 + lane%16]
__global__ void k_pack_b(const float* __restrict__ W, bf16_t* __restrict__ Wp,
                         int Kdim, int Ndim) {
    int tid  = blockIdx.x * blockDim.x + threadIdx.x;
    int nNT  = Ndim / 16;
    int nKT  = Kdim / 32;
    if (tid >= nKT * nNT * 32) return;
    int lane = tid & 31;
    int nt   = (tid >> 5) % nNT;
    int kt   = (tid >> 5) / nNT;
    int n    = nt * 16 + (lane & 15);
    int kh   = (lane >= 16) ? 16 : 0;
    bf16_t* dst = Wp + (size_t)tid * 16;
#pragma unroll
    for (int j = 0; j < 16; ++j) {
        int k = kt * 32 + (kh >> 1) + (j < 8 ? j : j + 8);
        dst[j] = (bf16_t)W[(size_t)k * Ndim + n];
    }
}

// ------------------------------------------------------------ GEMM1 (WMMA)
// h1[N,H] = relu(X[N,F] @ W1[F,H]).
// Block = 8 waves = 4 M-tiles x 2 N-halves. Each wave: 16 rows x 128 cols
// (8 v8f accumulators = 64 VGPRs -> no spills). The two waves sharing an
// M-tile issue identical X loads concurrently -> WGP$/L2 hits, X read ~once.
__global__ void k_gemm1(const float* __restrict__ X,
                        const bf16_t* __restrict__ W1p,
                        bf16_t* __restrict__ h1) {
    const int wave  = threadIdx.x >> 5;
    const int lane  = threadIdx.x & 31;
    const int mtile = blockIdx.x * 4 + (wave >> 1);
    const int nhalf = wave & 1;              // 0: cols 0..127, 1: cols 128..255
    if (mtile >= N_NODES / 16) return;

    const int m   = lane & 15;
    const int hi8 = (lane >= 16) ? 8 : 0;
    const float* row = X + (size_t)(mtile * 16 + m) * F_IN;
    const v16bf* Bp  = (const v16bf*)W1p;

    v8f zero;
#pragma unroll
    for (int i = 0; i < 8; ++i) zero[i] = 0.0f;
    v8f acc[8];
#pragma unroll
    for (int nt = 0; nt < 8; ++nt) acc[nt] = zero;

    for (int kt = 0; kt < F_IN / 32; ++kt) {
        // A operand: 16x32 bf16; lane holds row m, K = two contiguous chunks
        const float* c0 = row + kt * 32 + hi8;
        float4 q0 = *(const float4*)(c0);
        float4 q1 = *(const float4*)(c0 + 4);
        float4 q2 = *(const float4*)(c0 + 16);
        float4 q3 = *(const float4*)(c0 + 20);
        v16bf a;
        a[0]=(bf16_t)q0.x;  a[1]=(bf16_t)q0.y;  a[2]=(bf16_t)q0.z;  a[3]=(bf16_t)q0.w;
        a[4]=(bf16_t)q1.x;  a[5]=(bf16_t)q1.y;  a[6]=(bf16_t)q1.z;  a[7]=(bf16_t)q1.w;
        a[8]=(bf16_t)q2.x;  a[9]=(bf16_t)q2.y;  a[10]=(bf16_t)q2.z; a[11]=(bf16_t)q2.w;
        a[12]=(bf16_t)q3.x; a[13]=(bf16_t)q3.y; a[14]=(bf16_t)q3.z; a[15]=(bf16_t)q3.w;
#pragma unroll
        for (int nt = 0; nt < 8; ++nt) {
            v16bf b = Bp[(size_t)(kt * 16 + nhalf * 8 + nt) * 32 + lane];
            acc[nt] = __builtin_amdgcn_wmma_f32_16x16x32_bf16(
                false, a, false, b, (short)0, acc[nt], false, false);
        }
    }
    // D layout: lane L, vgpr r -> (M = r + (L>=16?8:0), Ncol = L%16). relu + bf16.
#pragma unroll
    for (int nt = 0; nt < 8; ++nt) {
#pragma unroll
        for (int r = 0; r < 8; ++r) {
            int mm = mtile * 16 + r + hi8;
            h1[(size_t)mm * H_DIM + (nhalf * 8 + nt) * 16 + m] =
                (bf16_t)fmaxf(acc[nt][r], 0.0f);
        }
    }
}

// ------------------------------------------------------------ GEMM2 (WMMA)
// h2[N,C] = h1[N,H] @ W2[H,C]; one wave per M-tile, 4 N-tiles (32 acc VGPRs).
__global__ void k_gemm2(const bf16_t* __restrict__ h1,
                        const bf16_t* __restrict__ W2p,
                        float* __restrict__ h2) {
    const int wave  = threadIdx.x >> 5;
    const int lane  = threadIdx.x & 31;
    const int mtile = blockIdx.x * 8 + wave;
    if (mtile >= N_NODES / 16) return;

    const int m   = lane & 15;
    const int hi8 = (lane >= 16) ? 8 : 0;
    const bf16_t* row = h1 + (size_t)(mtile * 16 + m) * H_DIM;
    const v16bf* Bp   = (const v16bf*)W2p;

    v8f zero;
#pragma unroll
    for (int i = 0; i < 8; ++i) zero[i] = 0.0f;
    v8f acc[4];
#pragma unroll
    for (int nt = 0; nt < 4; ++nt) acc[nt] = zero;

#pragma unroll
    for (int kt = 0; kt < H_DIM / 32; ++kt) {
        const bf16_t* c0 = row + kt * 32 + hi8;
        v8bf x0 = *(const v8bf*)(c0);
        v8bf x1 = *(const v8bf*)(c0 + 16);
        v16bf a;
#pragma unroll
        for (int j = 0; j < 8; ++j) { a[j] = x0[j]; a[j + 8] = x1[j]; }
#pragma unroll
        for (int nt = 0; nt < 4; ++nt) {
            v16bf b = Bp[(size_t)(kt * 4 + nt) * 32 + lane];
            acc[nt] = __builtin_amdgcn_wmma_f32_16x16x32_bf16(
                false, a, false, b, (short)0, acc[nt], false, false);
        }
    }
#pragma unroll
    for (int nt = 0; nt < 4; ++nt) {
#pragma unroll
        for (int r = 0; r < 8; ++r) {
            int mm = mtile * 16 + r + hi8;
            h2[(size_t)mm * C_DIM + nt * 16 + m] = acc[nt][r];
        }
    }
}

// ------------------------------------------------------------ propagation
__global__ void k_init_pnext(const float* __restrict__ h2,
                             float* __restrict__ pn, long n) {
    long i = (long)blockIdx.x * blockDim.x + threadIdx.x;
    if (i < n) pn[i] = ALPHA_F * h2[i];
}

// one wave per edge: pn[r,:] += (1-a)*w * p[c,:]   (C=64, 2 cols per lane)
// p/pn are 12.8 MB each -> L2-resident; gathers + fp32 atomics stay in L2.
__global__ void k_edge(const int* __restrict__ er, const int* __restrict__ ec,
                       const float* __restrict__ ew,
                       const float* __restrict__ p, float* __restrict__ pn) {
    int e = blockIdx.x * 8 + (threadIdx.x >> 5);
    if (e >= E_CNT) return;
    int lane = threadIdx.x & 31;
    int r = er[e], c = ec[e];
    float w = ew[e] * (1.0f - ALPHA_F);
    const float* src = p + (size_t)c * C_DIM;
    float* dst = pn + (size_t)r * C_DIM;
    atomicAdd(&dst[lane],      w * src[lane]);
    atomicAdd(&dst[lane + 32], w * src[lane + 32]);
}

// ------------------------------------------------------------ log_softmax
__global__ void k_logsoftmax(const float* __restrict__ p, float* __restrict__ out) {
    int row = blockIdx.x * 8 + (threadIdx.x >> 5);
    if (row >= N_NODES) return;
    int lane = threadIdx.x & 31;
    const float* x = p + (size_t)row * C_DIM;
    float a = x[lane], b = x[lane + 32];
    float mx = fmaxf(a, b);
#pragma unroll
    for (int off = 16; off >= 1; off >>= 1) mx = fmaxf(mx, __shfl_xor(mx, off, 32));
    float s = __expf(a - mx) + __expf(b - mx);
#pragma unroll
    for (int off = 16; off >= 1; off >>= 1) s += __shfl_xor(s, off, 32);
    float lse = mx + __logf(s);
    float* o = out + (size_t)row * C_DIM;
    o[lane]      = a - lse;
    o[lane + 32] = b - lse;
}

// ---------------------------------------------------------------- driver
extern "C" void kernel_launch(void* const* d_in, const int* in_sizes, int n_in,
                              void* d_out, int out_size, void* d_ws, size_t ws_size,
                              hipStream_t stream) {
    const int*   fidx = (const int*)d_in[0];     // [2, NNZ]: rows then cols
    const float* fval = (const float*)d_in[1];
    const int*   eidx = (const int*)d_in[2];     // [2, E]: rows then cols
    const float* ew   = (const float*)d_in[3];
    const float* W1   = (const float*)d_in[4];   // [F, H]
    const float* W2   = (const float*)d_in[5];   // [H, C]
    float* out = (float*)d_out;
    (void)in_sizes; (void)n_in; (void)out_size; (void)ws_size;

    char* ws = (char*)d_ws;
    // workspace layout (all offsets 256B-aligned)
    float*  X   = (float*)(ws + 0);                       // 204,800,000 B
    bf16_t* W1p = (bf16_t*)(ws + 204800000);              //     524,288 B
    bf16_t* W2p = (bf16_t*)(ws + 205324288);              //      32,768 B
    bf16_t* h1  = (bf16_t*)(ws + 205357056);              //  25,600,000 B
    float*  h2  = (float*)(ws + 230957056);               //  12,800,000 B
    float*  pA  = (float*)(ws + 243757056);               //  12,800,000 B
    float*  pB  = (float*)(ws + 256557056);               //  12,800,000 B

    const long nX = (long)N_NODES * F_IN;
    k_zero_f32<<<(unsigned)((nX + 255) / 256), 256, 0, stream>>>(X, nX);
    k_scatter_feat<<<(NNZ_CNT + 255) / 256, 256, 0, stream>>>(
        fidx, fidx + NNZ_CNT, fval, X, NNZ_CNT);

    k_pack_b<<<(32 * 16 * 32 + 255) / 256, 256, 0, stream>>>(W1, W1p, F_IN, H_DIM);
    k_pack_b<<<(8 * 4 * 32 + 255) / 256, 256, 0, stream>>>(W2, W2p, H_DIM, C_DIM);

    const int mtiles = N_NODES / 16;                 // 3125
    k_gemm1<<<(mtiles + 3) / 4, 256, 0, stream>>>(X, W1p, h1);
    k_gemm2<<<(mtiles + 7) / 8, 256, 0, stream>>>(h1, W2p, h2);

    const long nP = (long)N_NODES * C_DIM;
    const float* src = h2;
    float* dst = pA;
    for (int it = 0; it < K_ITERS; ++it) {
        k_init_pnext<<<(unsigned)((nP + 255) / 256), 256, 0, stream>>>(h2, dst, nP);
        k_edge<<<(E_CNT + 7) / 8, 256, 0, stream>>>(
            eidx, eidx + E_CNT, ew, src, dst);
        src = dst;
        dst = (dst == pA) ? pB : pA;
    }

    k_logsoftmax<<<(N_NODES + 7) / 8, 256, 0, stream>>>(src, out);
}
